// Pi05SuffixStepWrapper_8572754722964
// MI455X (gfx1250) — matrix operand order repaired and checked by reading, not compiled, not run on hardware
//
#include <hip/hip_runtime.h>
#include <hip/hip_bf16.h>
#include <math.h>

// ---------------------------------------------------------------------------
// Pi05 suffix step for MI455X (gfx1250, wave32, WMMA).
// All large GEMMs use v_wmma_f32_16x16x32_f16 (f16 in, f32 accumulate).
// LDS tiles are laid out so each lane's WMMA fragment is contiguous 16B runs
// => ds_load_b128; generic GEMM uses a 128x64 tile (2 M-frags per wave) for
// 1.5 ds_load_b128 per WMMA.
// ---------------------------------------------------------------------------

typedef _Float16 v16h __attribute__((ext_vector_type(16)));
typedef _Float16 h8   __attribute__((ext_vector_type(8)));
typedef _Float16 h4   __attribute__((ext_vector_type(4)));
typedef float    v8f  __attribute__((ext_vector_type(8)));

union FragU { v16h v; h8 h[2]; };

#define L_   12
#define B_   8
#define P_   800
#define D_   1024
#define H_   8
#define DH_  256
#define M_   4096
#define S_   50
#define A_   32
#define TK_  850            // P + S real keys
#define TP_  896            // padded key length (14 * 64)
#define NEG_ (-1e30f)

// A fragment (16x32 f16, ISA 7.12.2): lane (m, hi = lane>>4)
// halves e<8 -> k = e + 8*hi ; e>=8 -> k = e + 8 + 8*hi
// => two contiguous 8-half runs at [m][8*hi] and [m][16 + 8*hi].
__device__ __forceinline__ v16h frag_a(const _Float16* lsA, int m, int hi)
{
    FragU a;
    a.h[0] = *(const h8*)(lsA + m * 32 + 8 * hi);
    a.h[1] = *(const h8*)(lsA + m * 32 + 16 + 8 * hi);
    return a.v;
}

// B fragment: lane (n, hi): same k order as A (e<8 -> k = e + 8*hi; e>=8 -> k = e + 8 + 8*hi).
// LDS tile stored transposed [n][k] => two contiguous 8-half runs.
__device__ __forceinline__ v16h frag_b(const _Float16* lsBt, int n, int hi)
{
    FragU b;
    b.h[0] = *(const h8*)(lsBt + n * 32 + 8 * hi);
    b.h[1] = *(const h8*)(lsBt + n * 32 + 16 + 8 * hi);
    return b.v;
}

// Stage an nrows x 32 fp32 tile (row-major source, row stride lda) into LDS
// as f16 [row][k]. Assumes k0..k0+31 fully in-range (all K here are /32).
__device__ __forceinline__ void stage_rowmajor_tile(
    _Float16* ls, const float* src, int lda, int row0, int k0,
    int rowMax, int nrows, int tid)
{
    for (int i = tid; i < nrows * 8; i += 128) {    // 8 float4 per row
        int r = i >> 3;
        int c = (i & 7) << 2;
        float4 v = make_float4(0.f, 0.f, 0.f, 0.f);
        int gr = row0 + r;
        if (gr < rowMax)
            v = *(const float4*)(src + (size_t)gr * lda + k0 + c);
        h4 hv = { (_Float16)v.x, (_Float16)v.y, (_Float16)v.z, (_Float16)v.w };
        *(h4*)(ls + r * 32 + c) = hv;
    }
}

// ------------------------------ generic WMMA GEMM ---------------------------
// C[M,N] = beta*C + scale * act( A[M,K] @ B[K,N] + bias[N] )
// Tile 128x64, K-step 32; 4 waves, each wave owns 32 rows (2 M-frags).
// Requires: K % 32 == 0, N % 4 == 0 (true for every call here).
__global__ void __launch_bounds__(128)
gemm_wmma_kernel(const float* __restrict__ A, int lda,
                 const float* __restrict__ Bm, int ldb,
                 float* __restrict__ C, int ldc,
                 int Mr, int Nr, int Kr,
                 const float* __restrict__ bias,
                 int beta, int act, float scale)
{
    __shared__ __align__(16) _Float16 lsA[128 * 32];
    __shared__ __align__(16) _Float16 lsBt[64 * 32];   // transposed [n][k]
    const int tid  = threadIdx.x;
    const int lane = tid & 31;
    const int wave = tid >> 5;
    const int hi   = (lane >> 4) & 1;
    const int tileN = blockIdx.x * 64;
    const int tileM = blockIdx.y * 128;

    v8f acc[2][4] = {};

    for (int k0 = 0; k0 < Kr; k0 += 32) {
        // prefetch next K-chunk (global_prefetch_b8)
        if (k0 + 32 < Kr) {
            int gm = tileM + tid;
            if (gm < Mr) __builtin_prefetch(A + (size_t)gm * lda + k0 + 32, 0, 0);
            __builtin_prefetch(Bm + (size_t)(k0 + 32 + (tid >> 5)) * ldb + tileN, 0, 0);
        }
        stage_rowmajor_tile(lsA, A, lda, tileM, k0, Mr, 128, tid);
        // B tile: global [k][n] -> LDS transposed [n][k]
        for (int i = tid; i < 512; i += 128) {
            int k = i >> 4;                          // 0..31
            int n = (i & 15) << 2;                   // 0..60
            int gn = tileN + n;
            float4 v = make_float4(0.f, 0.f, 0.f, 0.f);
            if (gn < Nr)
                v = *(const float4*)(Bm + (size_t)(k0 + k) * ldb + gn);
            lsBt[(n + 0) * 32 + k] = (_Float16)v.x;
            lsBt[(n + 1) * 32 + k] = (_Float16)v.y;
            lsBt[(n + 2) * 32 + k] = (_Float16)v.z;
            lsBt[(n + 3) * 32 + k] = (_Float16)v.w;
        }
        __syncthreads();
        v16h a0 = frag_a(lsA, wave * 32 + (lane & 15), hi);
        v16h a1 = frag_a(lsA, wave * 32 + 16 + (lane & 15), hi);
#pragma unroll
        for (int nt = 0; nt < 4; ++nt) {
            v16h bf = frag_b(lsBt, nt * 16 + (lane & 15), hi);
            acc[0][nt] = __builtin_amdgcn_wmma_f32_16x16x32_f16(
                false, a0, false, bf, (short)0, acc[0][nt], false, false);
            acc[1][nt] = __builtin_amdgcn_wmma_f32_16x16x32_f16(
                false, a1, false, bf, (short)0, acc[1][nt], false, false);
        }
        __syncthreads();
    }

#pragma unroll
    for (int mf = 0; mf < 2; ++mf) {
#pragma unroll
        for (int nt = 0; nt < 4; ++nt) {
            int gn = tileN + nt * 16 + (lane & 15);
            if (gn >= Nr) continue;
            float bv = bias ? bias[gn] : 0.f;
#pragma unroll
            for (int r = 0; r < 8; ++r) {
                int gm = tileM + wave * 32 + mf * 16 + r + 8 * hi;
                if (gm >= Mr) continue;
                float v = acc[mf][nt][r] + bv;
                if (act == 1) {                       // silu
                    v = v / (1.f + __expf(-v));
                } else if (act == 2) {                // gelu (tanh approx)
                    float x3 = v * v * v;
                    v = 0.5f * v * (1.f + tanhf(0.7978845608f * (v + 0.044715f * x3)));
                }
                v *= scale;
                float* p = C + (size_t)gm * ldc + gn;
                *p = beta ? (*p + v) : v;
            }
        }
    }
}

// --------------------------- attention scores (WMMA) ------------------------
// scores[b,h,s,t] = (q[b,s,h,:] . key[b,t,:]) / 16 + mask bias.
// Keys are row-major [t][dh] in memory == the transposed-B layout directly.
__global__ void __launch_bounds__(128)
attn_scores_kernel(const float* __restrict__ q,       // [B,S,H,DH]
                   const float* __restrict__ cacheK,  // [B,P,DH] (layer slice)
                   const float* __restrict__ knew,    // [B,S,DH]
                   const unsigned char* __restrict__ pad, // [B,P]
                   float* __restrict__ scores)        // [B*H, S, TP_]
{
    const int bh = blockIdx.z;
    const int b  = bh >> 3;
    const int h  = bh & 7;
    const int tileN = blockIdx.x * 64;

    __shared__ __align__(16) _Float16 lsA[64 * 32];
    __shared__ __align__(16) _Float16 lsBt[64 * 32];   // [key][dh-chunk]
    const int tid  = threadIdx.x;
    const int lane = tid & 31;
    const int wave = tid >> 5;
    const int hi   = (lane >> 4) & 1;

    const float* qbh = q + ((size_t)b * S_ * H_ + h) * DH_;

    v8f acc[4] = {};

    for (int k0 = 0; k0 < DH_; k0 += 32) {
        stage_rowmajor_tile(lsA, qbh, H_ * DH_, 0, k0, S_, 64, tid);
        // key tile: rows are keys (n), cols dh (k) -- already [n][k] order
        for (int i = tid; i < 512; i += 128) {
            int nl = i >> 3;
            int c  = (i & 7) << 2;
            int gn = tileN + nl;
            float4 v = make_float4(0.f, 0.f, 0.f, 0.f);
            if (gn < P_)
                v = *(const float4*)(cacheK + ((size_t)b * P_ + gn) * DH_ + k0 + c);
            else if (gn < TK_)
                v = *(const float4*)(knew + ((size_t)b * S_ + (gn - P_)) * DH_ + k0 + c);
            h4 hv = { (_Float16)v.x, (_Float16)v.y, (_Float16)v.z, (_Float16)v.w };
            *(h4*)(lsBt + nl * 32 + c) = hv;
        }
        __syncthreads();
        v16h a = frag_a(lsA, wave * 16 + (lane & 15), hi);
#pragma unroll
        for (int nt = 0; nt < 4; ++nt) {
            v16h bf = frag_b(lsBt, nt * 16 + (lane & 15), hi);
            acc[nt] = __builtin_amdgcn_wmma_f32_16x16x32_f16(
                false, a, false, bf, (short)0, acc[nt], false, false);
        }
        __syncthreads();
    }

#pragma unroll
    for (int nt = 0; nt < 4; ++nt) {
        int gn = tileN + nt * 16 + (lane & 15);
        float biasv;
        if (gn < P_)       biasv = pad[(size_t)b * P_ + gn] ? 0.f : NEG_;
        else if (gn < TK_) biasv = 0.f;   // suffix att2d: cumsum pattern => all-visible
        else               biasv = NEG_;  // padding tail
#pragma unroll
        for (int r = 0; r < 8; ++r) {
            int s = wave * 16 + r + 8 * hi;
            if (s < S_)
                scores[((size_t)bh * S_ + s) * TP_ + gn] = acc[nt][r] * 0.0625f + biasv;
        }
    }
}

// ----------------------------- softmax over keys ----------------------------
__global__ void __launch_bounds__(128)
softmax_kernel(float* __restrict__ scores)   // rows of length TP_
{
    __shared__ float red[128];
    const int tid = threadIdx.x;
    float* p = scores + (size_t)blockIdx.x * TP_;

    float m = -3.4e38f;
    for (int j = tid; j < TP_; j += 128) m = fmaxf(m, p[j]);
    red[tid] = m; __syncthreads();
    for (int off = 64; off > 0; off >>= 1) {
        if (tid < off) red[tid] = fmaxf(red[tid], red[tid + off]);
        __syncthreads();
    }
    const float mx = red[0];
    __syncthreads();

    float s = 0.f;
    for (int j = tid; j < TP_; j += 128) {
        float e = __expf(p[j] - mx);
        p[j] = e; s += e;
    }
    red[tid] = s; __syncthreads();
    for (int off = 64; off > 0; off >>= 1) {
        if (tid < off) red[tid] += red[tid + off];
        __syncthreads();
    }
    const float inv = 1.f / red[0];
    for (int j = tid; j < TP_; j += 128) p[j] *= inv;
}

// ------------------------- attention output (WMMA) --------------------------
// o[b,s,h,:] = softmax_row @ vfull  (vfull from cache_v ++ fresh v)
__global__ void __launch_bounds__(128)
attn_out_kernel(const float* __restrict__ scores,  // [B*H, S, TP_]
                const float* __restrict__ cacheV,  // [B,P,DH]
                const float* __restrict__ vnew,    // [B,S,DH]
                float* __restrict__ o)             // [B,S,H,DH]
{
    const int bh = blockIdx.z;
    const int b  = bh >> 3;
    const int h  = bh & 7;
    const int tileN = blockIdx.x * 64;   // dh tile, 4 tiles

    __shared__ __align__(16) _Float16 lsA[64 * 32];
    __shared__ __align__(16) _Float16 lsBt[64 * 32];   // [dh][t-chunk]
    const int tid  = threadIdx.x;
    const int lane = tid & 31;
    const int wave = tid >> 5;
    const int hi   = (lane >> 4) & 1;

    const float* arow = scores + (size_t)bh * S_ * TP_;

    v8f acc[4] = {};

    for (int t0 = 0; t0 < TP_; t0 += 32) {
        stage_rowmajor_tile(lsA, arow, TP_, 0, t0, S_, 64, tid);
        // V tile: global [t][dh] -> LDS transposed [dh][t]
        for (int i = tid; i < 512; i += 128) {
            int k = i >> 4;                          // t offset 0..31
            int n = (i & 15) << 2;                   // dh offset
            int t = t0 + k;
            float4 v = make_float4(0.f, 0.f, 0.f, 0.f);
            if (t < P_)
                v = *(const float4*)(cacheV + ((size_t)b * P_ + t) * DH_ + tileN + n);
            else if (t < TK_)
                v = *(const float4*)(vnew + ((size_t)b * S_ + (t - P_)) * DH_ + tileN + n);
            lsBt[(n + 0) * 32 + k] = (_Float16)v.x;
            lsBt[(n + 1) * 32 + k] = (_Float16)v.y;
            lsBt[(n + 2) * 32 + k] = (_Float16)v.z;
            lsBt[(n + 3) * 32 + k] = (_Float16)v.w;
        }
        __syncthreads();
        v16h a = frag_a(lsA, wave * 16 + (lane & 15), hi);
#pragma unroll
        for (int nt = 0; nt < 4; ++nt) {
            v16h bf = frag_b(lsBt, nt * 16 + (lane & 15), hi);
            acc[nt] = __builtin_amdgcn_wmma_f32_16x16x32_f16(
                false, a, false, bf, (short)0, acc[nt], false, false);
        }
        __syncthreads();
    }

#pragma unroll
    for (int nt = 0; nt < 4; ++nt) {
        int gd = tileN + nt * 16 + (lane & 15);
#pragma unroll
        for (int r = 0; r < 8; ++r) {
            int s = wave * 16 + r + 8 * hi;
            if (s < S_)
                o[(((size_t)b * S_ + s) * H_ + h) * DH_ + gd] = acc[nt][r];
        }
    }
}

// ----------------------------- RMSNorm (+adaRMS) ----------------------------
__global__ void __launch_bounds__(256)
rmsnorm_kernel(const float* __restrict__ hin, const float* __restrict__ w,
               const float* __restrict__ condA, float* __restrict__ xout,
               int useAda)
{
    __shared__ float red[256];
    const int row = blockIdx.x;            // b*S + s
    const int b   = row / S_;
    const float* x = hin + (size_t)row * D_;

    float s = 0.f;
    for (int d = threadIdx.x; d < D_; d += 256) { float v = x[d]; s += v * v; }
    red[threadIdx.x] = s; __syncthreads();
    for (int off = 128; off > 0; off >>= 1) {
        if (threadIdx.x < off) red[threadIdx.x] += red[threadIdx.x + off];
        __syncthreads();
    }
    const float rs = rsqrtf(red[0] / (float)D_ + 1e-6f);
    for (int d = threadIdx.x; d < D_; d += 256) {
        float v = x[d] * rs * (1.f + w[d]);
        if (useAda) v *= (1.f + condA[(size_t)b * D_ + d]);
        xout[(size_t)row * D_ + d] = v;
    }
}

// --------------------------------- RoPE -------------------------------------
__global__ void rope_kernel(float* __restrict__ x, const float* __restrict__ cosb,
                            const float* __restrict__ sinb, int nheads)
{
    const int total = B_ * S_ * 128;
    int i = blockIdx.x * blockDim.x + threadIdx.x;
    if (i >= total * nheads) return;
    int j  = i & 127;
    int t  = i >> 7;          // (b*S+s)*nheads + hh
    int hh = t % nheads;
    int bs = t / nheads;
    float c  = cosb[bs * 128 + j];
    float sn = sinb[bs * 128 + j];
    float* p = x + ((size_t)bs * nheads + hh) * DH_;
    float x1 = p[j], x2 = p[j + 128];
    p[j]       = x1 * c - x2 * sn;
    p[j + 128] = x2 * c + x1 * sn;
}

__global__ void rope_cs_kernel(const float* __restrict__ offs,
                               float* __restrict__ cosb, float* __restrict__ sinb)
{
    int i = blockIdx.x * blockDim.x + threadIdx.x;
    if (i >= B_ * S_ * 128) return;
    int j  = i & 127;
    int bs = i >> 7;
    int b  = bs / S_;
    int s  = bs % S_;
    float inv = __expf(-logf(10000.f) * (2.f * (float)j) / (float)DH_);
    float ang = (offs[b] + (float)s) * inv;
    cosb[i] = cosf(ang);
    sinb[i] = sinf(ang);
}

__global__ void offs_kernel(const unsigned char* __restrict__ pad,
                            float* __restrict__ offs)
{
    int b = threadIdx.x;
    if (b >= B_) return;
    int c = 0;
    for (int j = 0; j < P_; ++j) c += (pad[(size_t)b * P_ + j] != 0);
    offs[b] = (float)c;
}

// ---------------------------- time embedding --------------------------------
__global__ void temb_kernel(const float* __restrict__ timestep, float* __restrict__ temb)
{
    int i = blockIdx.x * blockDim.x + threadIdx.x;
    if (i >= B_ * D_) return;
    int b = i >> 10;
    int d = i & 1023;
    int idx = (d < 512) ? d : d - 512;
    float f = __expf(-logf(10000.f) * (float)idx / 512.f);
    float a = timestep[b] * f;
    temb[i] = (d < 512) ? sinf(a) : cosf(a);
}

// ------------------------------ elementwise ---------------------------------
__global__ void mul_kernel(float* __restrict__ g, const float* __restrict__ u, int n)
{
    int i = blockIdx.x * blockDim.x + threadIdx.x;
    if (i < n) g[i] *= u[i];
}

// ------------------------------ final Euler step ----------------------------
__global__ void outproj_kernel(const float* __restrict__ hf,
                               const float* __restrict__ w,   // [D, A]
                               const float* __restrict__ bias,
                               const float* __restrict__ x_t,
                               const float* __restrict__ dt,
                               float* __restrict__ out)
{
    int i = blockIdx.x * blockDim.x + threadIdx.x;
    if (i >= B_ * S_ * A_) return;
    int a   = i & 31;
    int row = i >> 5;
    float s = bias[a];
    const float* hr = hf + (size_t)row * D_;
    for (int d = 0; d < D_; ++d) s += hr[d] * w[(size_t)d * A_ + a];
    out[i] = x_t[i] + dt[0] * s;
}

// ---------------------------------------------------------------------------
static inline void launch_gemm(const float* A, int lda, const float* B, int ldb,
                               float* C, int ldc, int M, int N, int K,
                               const float* bias, int beta, int act, float scale,
                               hipStream_t stream)
{
    dim3 grid((N + 63) / 64, (M + 127) / 128);
    gemm_wmma_kernel<<<grid, 128, 0, stream>>>(A, lda, B, ldb, C, ldc,
                                               M, N, K, bias, beta, act, scale);
}

extern "C" void kernel_launch(void* const* d_in, const int* in_sizes, int n_in,
                              void* d_out, int out_size, void* d_ws, size_t ws_size,
                              hipStream_t stream)
{
    const unsigned char* pad = (const unsigned char*)d_in[0];
    const float* x_t      = (const float*)d_in[1];
    const float* timestep = (const float*)d_in[2];
    const float* dt       = (const float*)d_in[3];
    const float* cache_k  = (const float*)d_in[4];
    const float* cache_v  = (const float*)d_in[5];
    const float* w_act_in = (const float*)d_in[6];
    const float* b_act_in = (const float*)d_in[7];
    const float* w_t1     = (const float*)d_in[8];
    const float* b_t1     = (const float*)d_in[9];
    const float* w_t2     = (const float*)d_in[10];
    const float* b_t2     = (const float*)d_in[11];
    const float* ln1_w    = (const float*)d_in[12];
    const float* ada1_w   = (const float*)d_in[13];
    const float* ln2_w    = (const float*)d_in[14];
    const float* wq       = (const float*)d_in[15];
    const float* wk       = (const float*)d_in[16];
    const float* wv       = (const float*)d_in[17];
    const float* wo       = (const float*)d_in[18];
    const float* wg       = (const float*)d_in[19];
    const float* wu       = (const float*)d_in[20];
    const float* wd       = (const float*)d_in[21];
    const float* lnf_w    = (const float*)d_in[22];
    const float* adaf_w   = (const float*)d_in[23];
    const float* w_act_out= (const float*)d_in[24];
    const float* b_act_out= (const float*)d_in[25];
    float* out = (float*)d_out;
    float* ws  = (float*)d_ws;

    const int BS = B_ * S_;            // 400
    // -------- workspace layout (floats); scores aliases g/u region --------
    float* h      = ws;                                  // 400*1024
    float* xb     = h      + (size_t)BS * D_;            // 400*1024
    float* q      = xb     + (size_t)BS * D_;            // 400*2048
    float* o      = q      + (size_t)BS * H_ * DH_;      // 400*2048
    float* kbuf   = o      + (size_t)BS * H_ * DH_;      // 400*256
    float* vbuf   = kbuf   + (size_t)BS * DH_;           // 400*256
    float* big    = vbuf   + (size_t)BS * DH_;           // max(scores, g+u)
    float* scores = big;                                 // 64*50*896
    float* g      = big;                                 // 400*4096 (aliases scores)
    float* u      = big    + (size_t)BS * M_;            // 400*4096
    float* cond   = big    + 2 * (size_t)BS * M_;        // 8*1024
    float* condA  = cond   + (size_t)B_ * D_;            // 8*1024
    float* temb   = condA  + (size_t)B_ * D_;            // 8*1024
    float* t1     = temb   + (size_t)B_ * D_;            // 8*1024
    float* cosb   = t1     + (size_t)B_ * D_;            // 8*50*128
    float* sinb   = cosb   + (size_t)B_ * S_ * 128;      // 8*50*128
    float* offs   = sinb   + (size_t)B_ * S_ * 128;      // 8

    // -------------------- prologue: positions, RoPE tables, cond ----------
    offs_kernel<<<1, 32, 0, stream>>>(pad, offs);
    rope_cs_kernel<<<(B_ * S_ * 128 + 255) / 256, 256, 0, stream>>>(offs, cosb, sinb);
    temb_kernel<<<(B_ * D_ + 255) / 256, 256, 0, stream>>>(timestep, temb);
    // cond = silu(temb @ w_t1 + b_t1) @ w_t2 + b_t2
    launch_gemm(temb, D_, w_t1, D_, t1,   D_, B_, D_, D_, b_t1, 0, 1, 1.f, stream);
    launch_gemm(t1,   D_, w_t2, D_, cond, D_, B_, D_, D_, b_t2, 0, 0, 1.f, stream);
    // h = (x_t @ w_act_in + b_act_in) * sqrt(D)
    launch_gemm(x_t, A_, w_act_in, D_, h, D_, BS, D_, A_, b_act_in, 0, 0, 32.f, stream);

    // ------------------------------ layers --------------------------------
    for (int l = 0; l < L_; ++l) {
        const float* l1 = ln1_w + (size_t)l * D_;
        const float* a1 = ada1_w + (size_t)l * D_ * D_;
        const float* l2 = ln2_w + (size_t)l * D_;
        const float* Wq = wq + (size_t)l * D_ * (H_ * DH_);
        const float* Wk = wk + (size_t)l * D_ * DH_;
        const float* Wv = wv + (size_t)l * D_ * DH_;
        const float* Wo = wo + (size_t)l * (H_ * DH_) * D_;
        const float* Wg = wg + (size_t)l * D_ * M_;
        const float* Wu = wu + (size_t)l * D_ * M_;
        const float* Wd = wd + (size_t)l * M_ * D_;
        const float* cK = cache_k + (size_t)l * B_ * P_ * DH_;
        const float* cV = cache_v + (size_t)l * B_ * P_ * DH_;

        // x = rms(h, l1) * (1 + cond @ a1)
        launch_gemm(cond, D_, a1, D_, condA, D_, B_, D_, D_, nullptr, 0, 0, 1.f, stream);
        rmsnorm_kernel<<<BS, 256, 0, stream>>>(h, l1, condA, xb, 1);

        // qkv projections
        launch_gemm(xb, D_, Wq, H_ * DH_, q,    H_ * DH_, BS, H_ * DH_, D_, nullptr, 0, 0, 1.f, stream);
        launch_gemm(xb, D_, Wk, DH_,      kbuf, DH_,      BS, DH_,      D_, nullptr, 0, 0, 1.f, stream);
        launch_gemm(xb, D_, Wv, DH_,      vbuf, DH_,      BS, DH_,      D_, nullptr, 0, 0, 1.f, stream);

        // RoPE on q and k
        rope_kernel<<<(B_ * S_ * H_ * 128 + 255) / 256, 256, 0, stream>>>(q, cosb, sinb, H_);
        rope_kernel<<<(B_ * S_ * 1 * 128 + 255) / 256, 256, 0, stream>>>(kbuf, cosb, sinb, 1);

        // attention: scores -> softmax -> output
        attn_scores_kernel<<<dim3(TP_ / 64, 1, B_ * H_), 128, 0, stream>>>(q, cK, kbuf, pad, scores);
        softmax_kernel<<<B_ * H_ * S_, 128, 0, stream>>>(scores);
        attn_out_kernel<<<dim3(DH_ / 64, 1, B_ * H_), 128, 0, stream>>>(scores, cV, vbuf, o);

        // h += o @ Wo
        launch_gemm(o, H_ * DH_, Wo, D_, h, D_, BS, D_, H_ * DH_, nullptr, 1, 0, 1.f, stream);

        // MLP: h += (gelu(x2@Wg) * (x2@Wu)) @ Wd
        rmsnorm_kernel<<<BS, 256, 0, stream>>>(h, l2, nullptr, xb, 0);
        launch_gemm(xb, D_, Wg, M_, g, M_, BS, M_, D_, nullptr, 0, 2, 1.f, stream);
        launch_gemm(xb, D_, Wu, M_, u, M_, BS, M_, D_, nullptr, 0, 0, 1.f, stream);
        mul_kernel<<<(BS * M_ + 255) / 256, 256, 0, stream>>>(g, u, BS * M_);
        launch_gemm(g, M_, Wd, D_, h, D_, BS, D_, M_, nullptr, 1, 0, 1.f, stream);
    }

    // ------------------------------ epilogue -------------------------------
    launch_gemm(cond, D_, adaf_w, D_, condA, D_, B_, D_, D_, nullptr, 0, 0, 1.f, stream);
    rmsnorm_kernel<<<BS, 256, 0, stream>>>(h, lnf_w, condA, xb, 1);
    outproj_kernel<<<(BS * A_ + 127) / 128, 128, 0, stream>>>(xb, w_act_out, b_act_out,
                                                              x_t, dt, out);
    (void)in_sizes; (void)n_in; (void)out_size; (void)ws_size;
}
